// TSHDnet_65274912964721
// MI455X (gfx1250) — matrix operand order repaired and checked by reading, not compiled, not run on hardware
//
#include <hip/hip_runtime.h>
#include <hip/hip_bf16.h>
#include <cmath>

// ---------------------------------------------------------------------------
// MI455X / gfx1250: wave32, WMMA 16x16x32 bf16 with f32 accumulate.
// All large contractions go through one tiled WMMA GEMM; small strided
// contractions are VALU kernels. Adjacency propagation is done as dense
// 1024x1024 bf16 GEMMs (the 2MB A matrix lives in L2 across all 8 props).
// LDS tiles are stored in fragment order so fragment loads are ds_load_b128.
// ---------------------------------------------------------------------------

typedef __bf16 bf16;
typedef __attribute__((ext_vector_type(16))) __bf16 v16bf;
typedef __attribute__((ext_vector_type(8)))  float  v8f;

__device__ __forceinline__ float b2f(bf16 x) { return (float)x; }
__device__ __forceinline__ bf16  f2b(float x) { return (bf16)x; }

// ------------------------- problem constants -------------------------------
#define NB 8
#define NT 12
#define NNODE 1024
#define NH 64
#define PTS (NB * NT * NNODE)          // 98304  (b,t,n) points
#define FEATF (NB * 32 * NH)           // 16384  node-major feature width
#define ROWS32 (NB * 32 * NNODE)       // 262144 rows for Wtf GEMMs

// ===========================================================================
// Generic tiled bf16 WMMA GEMM:  C[M,N] = act( A[M,K] x B[K,N] + bias
//                                            ; optional amix*Xmix + bmix*acc )
// A, B, Xmix row-major bf16. Writes fp32 Cf and/or bf16 Cb (null-skippable).
// Block: 128 threads = 4 waves; tile 64(M) x 64(N); K stepped by 32.
//
// LDS layouts (key optimization):
//  * As[r][k] holds the A tile unpermuted (contiguous k, same order as Bt),
//    so lane (half,m)'s 16-element fragment = 32 contiguous
//    bytes at As[row][half*16]. Global 8-elem chunks are copied unchanged
//    into the same aligned 8-elem slots, keeping b128 staging.
//  * Bt[n][k] is the B tile transposed; lane (half,n)'s fragment = K
//    half*16..half*16+15 at fixed column = 32 contiguous bytes.
// ===========================================================================
__global__ __launch_bounds__(128)
void gemm_bf16_kernel(const bf16* __restrict__ A, const bf16* __restrict__ Bm,
                      const float* __restrict__ bias, const bf16* __restrict__ Xmix,
                      float amix, float bmix, int M, int N, int K, int act,
                      float* __restrict__ Cf, bf16* __restrict__ Cb)
{
    __shared__ __align__(32) bf16 As[64][32];
    __shared__ __align__(32) bf16 Bt[64][32];
    const int tileN = blockIdx.x * 64;
    const int tileM = blockIdx.y * 64;
    const int t = threadIdx.x;
    const int wave = t >> 5, lane = t & 31;
    const int half = lane >> 4, l16 = lane & 15;

    v8f acc[4] = {};

    const bool fullM = (tileM + 64 <= M);
    const bool fullN = (tileN + 64 <= N);

    for (int kk = 0; kk < K; kk += 32) {
        const bool fullK = (kk + 32 <= K);

        // ---- stage A (64x32) into contiguous-k layout ----
        if (fullM && fullK) {
            int r = t >> 1;
            #pragma unroll
            for (int u = 0; u < 2; ++u) {
                int g  = (t & 1) * 2 + u;              // 8-column group 0..3
                int gp = g;                            // contiguous k (matches Bt)
                uint4 val = *(const uint4*)&A[(size_t)(tileM + r) * K + kk + g * 8];
                *(uint4*)&As[r][gp * 8] = val;
            }
        } else {
            for (int i = 0; i < 16; ++i) {
                int e = t + i * 128;
                int r = e >> 5, c = e & 31;
                int gr = tileM + r, gc = kk + c;
                int cp = c;
                As[r][cp] = (gr < M && gc < K) ? A[(size_t)gr * K + gc] : f2b(0.f);
            }
        }
        // ---- stage B (32x64) transposed into Bt[col][k] ----
        if (fullN && fullK) {
            #pragma unroll
            for (int u = 0; u < 2; ++u) {
                int chunk = t * 2 + u;                  // 256 chunks of 8 elems
                int r  = chunk >> 3;                    // k row 0..31
                int c0 = (chunk & 7) * 8;               // column group
                uint4 val = *(const uint4*)&Bm[(size_t)(kk + r) * N + tileN + c0];
                const bf16* vs = (const bf16*)&val;
                #pragma unroll
                for (int i = 0; i < 8; ++i) Bt[c0 + i][r] = vs[i];
            }
        } else {
            for (int i = 0; i < 16; ++i) {
                int e = t + i * 128;
                int r = e >> 6, c = e & 63;
                int gr = kk + r, gc = tileN + c;
                Bt[c][r] = (gr < K && gc < N) ? Bm[(size_t)gr * N + gc] : f2b(0.f);
            }
        }
        // prefetch next K-tile of A into L2 (global_prefetch_b8)
        if (kk + 32 < K && tileM + (t >> 1) < M)
            __builtin_prefetch(&A[(size_t)(tileM + (t >> 1)) * K + kk + 32], 0, 1);
        __syncthreads();

        // fragments: single 32B vector LDS load each -> ds_load_b128 pairs
        v16bf af = *(const v16bf*)&As[wave * 16 + l16][half * 16];
        #pragma unroll
        for (int j = 0; j < 4; ++j) {
            v16bf bfg = *(const v16bf*)&Bt[j * 16 + l16][half * 16];
            acc[j] = __builtin_amdgcn_wmma_f32_16x16x32_bf16(
                false, af, false, bfg, (short)0, acc[j], false, false);
        }
        __syncthreads();
    }

    // epilogue: C layout = VGPR r -> (M = half*8 + r), N = lane&15
    #pragma unroll
    for (int j = 0; j < 4; ++j) {
        #pragma unroll
        for (int r = 0; r < 8; ++r) {
            int row = tileM + wave * 16 + half * 8 + r;
            int col = tileN + j * 16 + l16;
            if (row < M && col < N) {
                float v = acc[j][r];
                if (bias) v += bias[col];
                if (Xmix) v = amix * b2f(Xmix[(size_t)row * N + col]) + bmix * v;
                if (act == 1) v = fmaxf(v, 0.f);
                else if (act == 2) v = 1.f / (1.f + __expf(-v));
                if (Cf) Cf[(size_t)row * N + col] = v;
                if (Cb) Cb[(size_t)row * N + col] = f2b(v);
            }
        }
    }
}

// ======================= graph construction kernels ========================

// n = tanh(3*(emb @ w + b)) : 1024 rows x 40
__global__ void node_feat_kernel(const float* __restrict__ emb, const float* __restrict__ w,
                                 const float* __restrict__ b, float* __restrict__ out)
{
    int row = blockIdx.x, j = threadIdx.x;
    if (j < 40) {
        float s = b[j];
        for (int k = 0; k < 40; ++k) s += emb[row * 40 + k] * w[k * 40 + j];
        out[row * 40 + j] = tanhf(3.0f * s);
    }
}

// adj = relu(tanh(3*(n1 n2^T - n2 n1^T)))
__global__ void adj_kernel(const float* __restrict__ n1, const float* __restrict__ n2,
                           float* __restrict__ adj)
{
    int j = blockIdx.x * 16 + threadIdx.x;
    int i = blockIdx.y * 16 + threadIdx.y;
    float d1 = 0.f, d2 = 0.f;
    for (int k = 0; k < 40; ++k) {
        d1 += n1[i * 40 + k] * n2[j * 40 + k];
        d2 += n2[i * 40 + k] * n1[j * 40 + k];
    }
    adj[i * 1024 + j] = fmaxf(tanhf(3.0f * (d1 - d2)), 0.f);
}

// per-row top-10 mask (ties -> lowest index, matching lax.top_k), plus rowsum
__global__ __launch_bounds__(256)
void topk_kernel(const float* __restrict__ adj, float* __restrict__ adjm,
                 float* __restrict__ rowsum)
{
    __shared__ float sv[256];
    __shared__ int   si[256];
    __shared__ int   selc[10];
    __shared__ float selv[10];
    int row = blockIdx.x, t = threadIdx.x;
    float v[4];
    for (int i = 0; i < 4; ++i) v[i] = adj[row * 1024 + t + 256 * i];

    for (int k = 0; k < 10; ++k) {
        float bv = -1.f; int bc = 0x7fffffff;
        for (int i = 0; i < 4; ++i)
            if (v[i] > bv) { bv = v[i]; bc = t + 256 * i; }
        sv[t] = bv; si[t] = bc;
        __syncthreads();
        for (int s = 128; s > 0; s >>= 1) {
            if (t < s) {
                if (sv[t + s] > sv[t] || (sv[t + s] == sv[t] && si[t + s] < si[t])) {
                    sv[t] = sv[t + s]; si[t] = si[t + s];
                }
            }
            __syncthreads();
        }
        if (t == 0) { selc[k] = si[0]; selv[k] = sv[0]; }
        __syncthreads();
        int wsel = selc[k];
        if ((wsel & 255) == t) v[wsel >> 8] = -1.f;   // exclude winner
        __syncthreads();
    }
    if (t == 0) {
        float rs = 1.f;                                // +identity
        for (int k = 0; k < 10; ++k) rs += selv[k];
        rowsum[row] = rs;
    }
    for (int i = 0; i < 4; ++i) {
        int col = t + 256 * i;
        bool keep = false;
        for (int k = 0; k < 10; ++k) keep |= (selc[k] == col);
        adjm[row * 1024 + col] = keep ? adj[row * 1024 + col] : 0.f;
    }
}

__global__ void colsum_kernel(const float* __restrict__ adjm, float* __restrict__ colsum)
{
    int col = blockIdx.x * 256 + threadIdx.x;
    float s = 1.f;
    for (int r = 0; r < 1024; ++r) s += adjm[r * 1024 + col];
    colsum[col] = s;
}

// AN = (adjm + I)/rowsum ; ATN = (adjm^T + I)/colsum  -> bf16
__global__ void norm_adj_kernel(const float* __restrict__ adjm, const float* __restrict__ rowsum,
                                const float* __restrict__ colsum, bf16* __restrict__ ANb,
                                bf16* __restrict__ ATNb)
{
    int idx = blockIdx.x * 256 + threadIdx.x;
    int wcol = idx & 1023, v = idx >> 10;
    float eye = (v == wcol) ? 1.f : 0.f;
    ANb[idx]  = f2b((adjm[idx] + eye) / rowsum[v]);
    ATNb[idx] = f2b((adjm[wcol * 1024 + v] + eye) / colsum[v]);
}

// ======================= embedding / gating kernels ========================

// per point: gather tid/diw embeddings, hd = feat @ W_emb + b, build g_in bf16
__global__ void embed_kernel(const float* __restrict__ hist, const float* __restrict__ nodeu,
                             const float* __restrict__ tde, const float* __restrict__ dwe,
                             const float* __restrict__ W_emb, const float* __restrict__ b_emb,
                             float* __restrict__ tid_g, float* __restrict__ diw_g,
                             float* __restrict__ hd, bf16* __restrict__ g_in)
{
    int p = blockIdx.x, h = threadIdx.x;                 // 98304 x 64
    const float* hp = hist + (size_t)p * 4;
    int ti = (int)(hp[2] * 288.f); ti = ti < 0 ? 0 : (ti > 287 ? 287 : ti);
    int di = (int)hp[3];           di = di < 0 ? 0 : (di > 6 ? 6 : di);
    float f0 = hp[0], f1 = hp[1];
    hd[(size_t)p * 64 + h] = f0 * W_emb[h] + f1 * W_emb[64 + h] + b_emb[h];
    int n = p & 1023;
    float g;
    if (h < 40)      g = nodeu[n * 40 + h];
    else if (h < 52) g = tde[ti * 12 + (h - 40)];
    else             g = dwe[di * 12 + (h - 52)];
    g_in[(size_t)p * 64 + h] = f2b(g);
    if (h < 12) {
        tid_g[(size_t)p * 12 + h] = tde[ti * 12 + h];
        diw_g[(size_t)p * 12 + h] = dwe[di * 12 + h];
    }
}

// x1 = gate*hd, x2 = hd - x1, gated = [x1|x2]
__global__ void gating_kernel(const float* __restrict__ gate, const float* __restrict__ hd,
                              float* __restrict__ gated, float* __restrict__ x1f,
                              float* __restrict__ x2f)
{
    int idx = blockIdx.x * 256 + threadIdx.x;            // PTS*64
    int h = idx & 63; size_t p = (size_t)(idx >> 6);
    float g = gate[idx], v = hd[idx];
    float a = g * v, b = v - a;
    x1f[idx] = a; x2f[idx] = b;
    gated[p * 128 + h] = a;
    gated[p * 128 + 64 + h] = b;
}

// einsum('btnh,ot->bonh', x, Wsc)+bsc, relu; write node-major bf16 (n, b*2048+o*64+h)
__global__ void wsc_kernel(const float* __restrict__ x, const float* __restrict__ Wsc,
                           const float* __restrict__ bsc, bf16* __restrict__ outp)
{
    int idx = blockIdx.x * 256 + threadIdx.x;            // 16.8M
    int h = idx & 63;
    int n = (idx >> 6) & 1023;
    int o = (idx >> 16) & 31;
    int b = idx >> 21;
    float s = bsc[o];
    for (int t = 0; t < 12; ++t)
        s += Wsc[o * 12 + t] * x[(((size_t)(b * 12 + t) * 1024 + n) << 6) + h];
    s = fmaxf(s, 0.f);
    outp[((size_t)n << 14) + (b << 11) + (o << 6) + h] = f2b(s);
}

// mixprop projections: H = proj(P0,P1n,P2n; w1)+proj(P0,P1t,P2t; w2); write into
// the concatenated (B,32,N,256) tensor at feature offset goff.
__global__ void mixproj_kernel(const bf16* __restrict__ P0, const bf16* __restrict__ P1n,
                               const bf16* __restrict__ P2n, const bf16* __restrict__ P1t,
                               const bf16* __restrict__ P2t, const float* __restrict__ w1,
                               const float* __restrict__ b1, const float* __restrict__ w2,
                               const float* __restrict__ b2, bf16* __restrict__ OUT, int goff)
{
    int idx = blockIdx.x * 256 + threadIdx.x;            // 16.8M
    int l = idx & 63;
    int n = (idx >> 6) & 1023;
    int o = (idx >> 16) & 31;
    int b = idx >> 21;
    size_t base = ((size_t)n << 14) + (b << 11) + l;
    float s = b1[o] + b2[o];
    for (int c = 0; c < 32; ++c) {
        size_t ix = base + (c << 6);
        s += (w1[o * 96 + c] + w2[o * 96 + c]) * b2f(P0[ix]);
        s += w1[o * 96 + 32 + c] * b2f(P1n[ix]);
        s += w1[o * 96 + 64 + c] * b2f(P2n[ix]);
        s += w2[o * 96 + 32 + c] * b2f(P1t[ix]);
        s += w2[o * 96 + 64 + c] * b2f(P2t[ix]);
    }
    OUT[(((size_t)(b * 32 + o) << 10) + n) * 256 + goff + l] = f2b(s);
}

// fused conv1x1 pair: 32 -> relu(128) -> 12 over channel axis of (B,32,N,256)
__global__ void ec_kernel(const float* __restrict__ XTF, const float* __restrict__ W1,
                          const float* __restrict__ bi1, const float* __restrict__ W2,
                          const float* __restrict__ bi2, float* __restrict__ xec)
{
    int idx = blockIdx.x * 256 + threadIdx.x;            // 2.1M
    int l = idx & 255;
    int n = (idx >> 8) & 1023;
    int b = idx >> 18;
    float xin[32];
    for (int c = 0; c < 32; ++c)
        xin[c] = XTF[(((size_t)(b * 32 + c) << 10) + n) * 256 + l];
    float acc[12];
    for (int o = 0; o < 12; ++o) acc[o] = bi2[o];
    for (int j = 0; j < 128; ++j) {
        float hsum = bi1[j];
        for (int c = 0; c < 32; ++c) hsum += W1[j * 32 + c] * xin[c];
        hsum = fmaxf(hsum, 0.f);
        for (int o = 0; o < 12; ++o) acc[o] += W2[o * 128 + j] * hsum;
    }
    for (int o = 0; o < 12; ++o)
        xec[(((size_t)(b * 12 + o) << 10) + n) * 256 + l] = acc[o];
}

// fh = relu(concat[gated(128) | xec(256) | tid(12) | diw(12) | hd(64)]) -> bf16
__global__ void fh_kernel(const float* __restrict__ gated, const float* __restrict__ xec,
                          const float* __restrict__ tid_g, const float* __restrict__ diw_g,
                          const float* __restrict__ hd, bf16* __restrict__ fhb)
{
    size_t p = blockIdx.x;                               // 98304 rows
    for (int col = threadIdx.x; col < 472; col += 128) {
        float v;
        if (col < 128)      v = gated[p * 128 + col];
        else if (col < 384) v = xec[p * 256 + (col - 128)];
        else if (col < 396) v = tid_g[p * 12 + (col - 384)];
        else if (col < 408) v = diw_g[p * 12 + (col - 396)];
        else                v = hd[p * 64 + (col - 408)];
        fhb[p * 472 + col] = f2b(fmaxf(v, 0.f));
    }
}

// Wec3 over T (12->4), then transpose (B,T,N,3)->(B,N,4,3) and flatten
__global__ void final_kernel(const float* __restrict__ Ff, const float* __restrict__ Wec3,
                             const float* __restrict__ bec3, float* __restrict__ out)
{
    int idx = blockIdx.x * 256 + threadIdx.x;            // 98304
    if (idx >= NB * NNODE * 12) return;
    int c3 = idx % 3;
    int o  = (idx / 3) & 3;
    int rest = idx / 12;
    int n = rest & 1023;
    int b = rest >> 10;
    float s = bec3[o];
    for (int t = 0; t < 12; ++t)
        s += Wec3[o * 12 + t] * Ff[(((size_t)(b * 12 + t) << 10) + n) * 3 + c3];
    out[((size_t)(b << 10) + n) * 12 + o * 3 + c3] = s;
}

__global__ void f2b_kernel(const float* __restrict__ s, bf16* __restrict__ d, int n)
{
    int i = blockIdx.x * 256 + threadIdx.x;
    if (i < n) d[i] = f2b(s[i]);
}

// ===========================================================================
extern "C" void kernel_launch(void* const* d_in, const int* in_sizes, int n_in,
                              void* d_out, int out_size, void* d_ws, size_t ws_size,
                              hipStream_t stream)
{
    (void)in_sizes; (void)n_in; (void)out_size; (void)ws_size;

    const float* hist  = (const float*)d_in[0];
    const float* nodeu = (const float*)d_in[1];
    const float* tde   = (const float*)d_in[2];
    const float* dwe   = (const float*)d_in[3];
    const float* gce1  = (const float*)d_in[4];
    const float* gce2  = (const float*)d_in[5];
    const float* gcw1  = (const float*)d_in[6];
    const float* gcb1  = (const float*)d_in[7];
    const float* gcw2  = (const float*)d_in[8];
    const float* gcb2  = (const float*)d_in[9];
    const float* W_emb = (const float*)d_in[10];
    const float* b_emb = (const float*)d_in[11];
    const float* Wg1   = (const float*)d_in[12];
    const float* bg1   = (const float*)d_in[13];
    const float* Wg2   = (const float*)d_in[14];
    const float* bg2   = (const float*)d_in[15];
    const float* Wsc   = (const float*)d_in[16];
    const float* bsc   = (const float*)d_in[17];
    const float* mix1w = (const float*)d_in[18];
    const float* mix1b = (const float*)d_in[19];
    const float* mix2w = (const float*)d_in[20];
    const float* mix2b = (const float*)d_in[21];
    const float* Wtf1  = (const float*)d_in[22];
    const float* btf1  = (const float*)d_in[23];
    const float* Wtf2  = (const float*)d_in[24];
    const float* btf2  = (const float*)d_in[25];
    const float* Wec1  = (const float*)d_in[26];
    const float* bec1  = (const float*)d_in[27];
    const float* Wec2  = (const float*)d_in[28];
    const float* bec2  = (const float*)d_in[29];
    const float* Wfc1  = (const float*)d_in[30];
    const float* bfc1  = (const float*)d_in[31];
    const float* Wfc2  = (const float*)d_in[32];
    const float* bfc2  = (const float*)d_in[33];
    const float* Wec3  = (const float*)d_in[34];
    const float* bec3  = (const float*)d_in[35];
    float* out = (float*)d_out;

    // ---------------- workspace carve (sequential, 256B aligned) ------------
    char* wsp = (char*)d_ws;
    auto alloc = [&](size_t bytes) -> void* {
        void* p = (void*)wsp; wsp += (bytes + 255) & ~(size_t)255; return p;
    };
    float* n1     = (float*)alloc(1024 * 40 * 4);
    float* n2     = (float*)alloc(1024 * 40 * 4);
    float* adj    = (float*)alloc((size_t)1024 * 1024 * 4);
    float* adjm   = (float*)alloc((size_t)1024 * 1024 * 4);
    float* rowsum = (float*)alloc(1024 * 4);
    float* colsum = (float*)alloc(1024 * 4);
    bf16*  ANb    = (bf16*)alloc((size_t)1024 * 1024 * 2);
    bf16*  ATNb   = (bf16*)alloc((size_t)1024 * 1024 * 2);
    float* tid_g  = (float*)alloc((size_t)PTS * 12 * 4);
    float* diw_g  = (float*)alloc((size_t)PTS * 12 * 4);
    float* hd     = (float*)alloc((size_t)PTS * 64 * 4);
    bf16*  g_in   = (bf16*)alloc((size_t)PTS * 64 * 2);
    bf16*  GH     = (bf16*)alloc((size_t)PTS * 64 * 2);
    float* gatef  = (float*)alloc((size_t)PTS * 64 * 4);
    float* gated  = (float*)alloc((size_t)PTS * 128 * 4);
    float* x1f    = (float*)alloc((size_t)PTS * 64 * 4);
    float* x2f    = (float*)alloc((size_t)PTS * 64 * 4);
    bf16*  X1p    = (bf16*)alloc((size_t)NNODE * FEATF * 2);
    bf16*  X2p    = (bf16*)alloc((size_t)NNODE * FEATF * 2);
    bf16*  P1N    = (bf16*)alloc((size_t)NNODE * FEATF * 2);
    bf16*  P2N    = (bf16*)alloc((size_t)NNODE * FEATF * 2);
    bf16*  P1T    = (bf16*)alloc((size_t)NNODE * FEATF * 2);
    bf16*  P2T    = (bf16*)alloc((size_t)NNODE * FEATF * 2);
    bf16*  OUTb   = (bf16*)alloc((size_t)ROWS32 * 256 * 2);
    bf16*  HIDb   = (bf16*)alloc((size_t)ROWS32 * 256 * 2);
    float* XTF    = (float*)alloc((size_t)ROWS32 * 256 * 4);
    float* xec    = (float*)alloc((size_t)NB * 12 * NNODE * 256 * 4);
    bf16*  fhb    = (bf16*)alloc((size_t)PTS * 472 * 2);
    bf16*  FH1    = (bf16*)alloc((size_t)PTS * 512 * 2);
    float* Ff     = (float*)alloc((size_t)PTS * 3 * 4);
    bf16*  Wg1b   = (bf16*)alloc(64 * 64 * 2);
    bf16*  Wg2b   = (bf16*)alloc(64 * 64 * 2);
    bf16*  Wtf1b  = (bf16*)alloc(256 * 256 * 2);
    bf16*  Wtf2b  = (bf16*)alloc(256 * 256 * 2);
    bf16*  Wfc1b  = (bf16*)alloc(472 * 512 * 2);
    bf16*  Wfc2b  = (bf16*)alloc(512 * 3 * 2);

    auto gemm = [&](const bf16* A, const bf16* B_, const float* bias, const bf16* Xm,
                    float amix, float bmix, int M, int Nn, int K, int act,
                    float* Cf, bf16* Cb) {
        dim3 g((Nn + 63) / 64, (M + 63) / 64);
        gemm_bf16_kernel<<<g, 128, 0, stream>>>(A, B_, bias, Xm, amix, bmix,
                                                M, Nn, K, act, Cf, Cb);
    };
    auto cvt = [&](const float* s, bf16* d, int n) {
        f2b_kernel<<<(n + 255) / 256, 256, 0, stream>>>(s, d, n);
    };

    // weight conversions to bf16
    cvt(Wg1, Wg1b, 64 * 64);
    cvt(Wg2, Wg2b, 64 * 64);
    cvt(Wtf1, Wtf1b, 256 * 256);
    cvt(Wtf2, Wtf2b, 256 * 256);
    cvt(Wfc1, Wfc1b, 472 * 512);
    cvt(Wfc2, Wfc2b, 512 * 3);

    // ---- graph construction ------------------------------------------------
    node_feat_kernel<<<1024, 64, 0, stream>>>(gce1, gcw1, gcb1, n1);
    node_feat_kernel<<<1024, 64, 0, stream>>>(gce2, gcw2, gcb2, n2);
    adj_kernel<<<dim3(64, 64), dim3(16, 16), 0, stream>>>(n1, n2, adj);
    topk_kernel<<<1024, 256, 0, stream>>>(adj, adjm, rowsum);
    colsum_kernel<<<4, 256, 0, stream>>>(adjm, colsum);
    norm_adj_kernel<<<(1024 * 1024) / 256, 256, 0, stream>>>(adjm, rowsum, colsum, ANb, ATNb);

    // ---- embeddings + gate MLP (WMMA) -------------------------------------
    embed_kernel<<<PTS, 64, 0, stream>>>(hist, nodeu, tde, dwe, W_emb, b_emb,
                                         tid_g, diw_g, hd, g_in);
    gemm(g_in, Wg1b, bg1, nullptr, 0.f, 1.f, PTS, 64, 64, /*relu*/1, nullptr, GH);
    gemm(GH, Wg2b, bg2, nullptr, 0.f, 1.f, PTS, 64, 64, /*sigmoid*/2, gatef, nullptr);
    gating_kernel<<<(PTS * 64) / 256, 256, 0, stream>>>(gatef, hd, gated, x1f, x2f);

    // ---- time contraction into node-major bf16 -----------------------------
    wsc_kernel<<<(NB * 32 * NNODE * NH) / 256, 256, 0, stream>>>(x1f, Wsc, bsc, X1p);
    wsc_kernel<<<(NB * 32 * NNODE * NH) / 256, 256, 0, stream>>>(x2f, Wsc, bsc, X2p);

    // ---- mixprop: dense propagation GEMMs with fused alpha-mix epilogue ----
    const float PA = 0.05f, PB = 0.95f;
    // x1 chains (shared across i=0,1)
    gemm(ANb,  X1p, nullptr, X1p, PA, PB, NNODE, FEATF, NNODE, 0, nullptr, P1N);
    gemm(ANb,  P1N, nullptr, X1p, PA, PB, NNODE, FEATF, NNODE, 0, nullptr, P2N);
    gemm(ATNb, X1p, nullptr, X1p, PA, PB, NNODE, FEATF, NNODE, 0, nullptr, P1T);
    gemm(ATNb, P1T, nullptr, X1p, PA, PB, NNODE, FEATF, NNODE, 0, nullptr, P2T);
    mixproj_kernel<<<(NB * 32 * NNODE * NH) / 256, 256, 0, stream>>>(
        X1p, P1N, P2N, P1T, P2T, mix1w, mix1b, mix2w, mix2b, OUTb, 0);        // H1,i=0
    mixproj_kernel<<<(NB * 32 * NNODE * NH) / 256, 256, 0, stream>>>(
        X1p, P1N, P2N, P1T, P2T, mix1w + 32 * 96, mix1b + 32,
        mix2w + 32 * 96, mix2b + 32, OUTb, 128);                              // H1,i=1
    // x2 chains (reuse P buffers)
    gemm(ANb,  X2p, nullptr, X2p, PA, PB, NNODE, FEATF, NNODE, 0, nullptr, P1N);
    gemm(ANb,  P1N, nullptr, X2p, PA, PB, NNODE, FEATF, NNODE, 0, nullptr, P2N);
    gemm(ATNb, X2p, nullptr, X2p, PA, PB, NNODE, FEATF, NNODE, 0, nullptr, P1T);
    gemm(ATNb, P1T, nullptr, X2p, PA, PB, NNODE, FEATF, NNODE, 0, nullptr, P2T);
    mixproj_kernel<<<(NB * 32 * NNODE * NH) / 256, 256, 0, stream>>>(
        X2p, P1N, P2N, P1T, P2T, mix1w, mix1b, mix2w, mix2b, OUTb, 64);       // H2,i=0
    mixproj_kernel<<<(NB * 32 * NNODE * NH) / 256, 256, 0, stream>>>(
        X2p, P1N, P2N, P1T, P2T, mix1w + 32 * 96, mix1b + 32,
        mix2w + 32 * 96, mix2b + 32, OUTb, 192);                              // H2,i=1

    // ---- temporal FFN (WMMA) ----------------------------------------------
    gemm(OUTb, Wtf1b, btf1, nullptr, 0.f, 1.f, ROWS32, 256, 256, 1, nullptr, HIDb);
    gemm(HIDb, Wtf2b, btf2, nullptr, 0.f, 1.f, ROWS32, 256, 256, 0, XTF, nullptr);

    // ---- Wec1/Wec2 fused ---------------------------------------------------
    ec_kernel<<<(NB * NNODE * 256) / 256, 256, 0, stream>>>(XTF, Wec1, bec1, Wec2, bec2, xec);

    // ---- fh concat + final MLP (WMMA) -------------------------------------
    fh_kernel<<<PTS, 128, 0, stream>>>(gated, xec, tid_g, diw_g, hd, fhb);
    gemm(fhb, Wfc1b, bfc1, nullptr, 0.f, 1.f, PTS, 512, 472, 1, nullptr, FH1);
    gemm(FH1, Wfc2b, bfc2, nullptr, 0.f, 1.f, PTS, 3, 512, 0, Ff, nullptr);

    // ---- Wec3 + transpose --------------------------------------------------
    final_kernel<<<(NB * NNODE * 12 + 255) / 256, 256, 0, stream>>>(Ff, Wec3, bec3, out);
}
